// DgcnNetwork_24670292149146
// MI455X (gfx1250) — matrix operand order repaired and checked by reading, not compiled, not run on hardware
//
#include <hip/hip_runtime.h>

#define EPS 1e-5f
typedef _Float16 h16;
typedef __attribute__((ext_vector_type(16))) _Float16 v16h;
typedef __attribute__((ext_vector_type(8)))  float    v8f;

static constexpr int Nn   = 20000;
static constexpr int Ee   = 640000;
static constexpr int Gb   = 32;
static constexpr int NCLS = 24;

union AFrag { v16h v; uint4 q[2]; };

// A fragment (16x32 f16, M x K): lane L -> M = L&15; halves = row[k0+hi*8 .. +7], row[k0+hi*8+16 .. +23]
__device__ inline v16h frag_a(const h16* row, int k0, int hi) {
  AFrag f;
  const h16* p = row + k0 + hi * 8;
  f.q[0] = *(const uint4*)(p);
  f.q[1] = *(const uint4*)(p + 16);
  return f.v;
}
// B fragment (32x16 f16, K x N) from W^T[N][K]: lane L -> N = L&15; halves = Wt[N][k0+hi*8 .. +7], Wt[N][k0+hi*8+16 .. +23]
__device__ inline v16h frag_b(const h16* row, int k0, int hi) {
  AFrag f;
  const h16* p = row + k0 + hi * 8;
  f.q[0] = *(const uint4*)(p);
  f.q[1] = *(const uint4*)(p + 16);
  return f.v;
}
#define WMMA(a, b, c) __builtin_amdgcn_wmma_f32_16x16x32_f16(false, (a), false, (b), (short)0, (c), false, false)
// Forbid the scheduler from sinking fragment loads back into the WMMA chain:
// everything before the barrier must issue first -> fragments stay in distinct
// VGPRs and the per-WMMA s_wait_dscnt 0 collapses into partial in-order waits.
#define SCHED_FENCE() __builtin_amdgcn_sched_barrier(0)

// ---------------------------------------------------------------- utilities
__global__ void k_fill(float* __restrict__ p, long long n, float v) {
  long long i = (long long)blockIdx.x * blockDim.x + threadIdx.x;
  long long st = (long long)gridDim.x * blockDim.x;
  for (; i < n; i += st) p[i] = v;
}

// Wt[n*K+k] = W[k*Nc+n]  (f32 -> f16 transpose, coalesced writes)
__global__ void k_convert_wt(const float* __restrict__ W, h16* __restrict__ Wt, int K, int Nc) {
  int total = K * Nc;
  for (int i = blockIdx.x * blockDim.x + threadIdx.x; i < total; i += gridDim.x * blockDim.x) {
    int n = i / K, k = i - n * K;
    Wt[i] = (h16)W[k * Nc + n];
  }
}

__global__ void k_count(const long long* __restrict__ batch, float* __restrict__ cnt, int n) {
  for (int i = blockIdx.x * blockDim.x + threadIdx.x; i < n; i += gridDim.x * blockDim.x)
    atomicAdd(&cnt[(int)batch[i]], 1.f);
}

// ------------------------------------------------- conv1 layer1 (K=6, scalar)
__global__ __launch_bounds__(128) void k_edge_conv1_l1(
    const float* __restrict__ x, const long long* __restrict__ src, const long long* __restrict__ dst,
    const float* __restrict__ W1, const float* __restrict__ b1,
    h16* __restrict__ h1, float* __restrict__ gsum, float* __restrict__ gsq)
{
  int c = threadIdx.x;
  float w0 = W1[c], w1 = W1[128 + c], w2 = W1[256 + c];
  float w3 = W1[384 + c], w4 = W1[512 + c], w5 = W1[640 + c];
  float bias = b1[c], s1 = 0.f, s2 = 0.f;
  int e0 = blockIdx.x * 512;
  for (int i = 0; i < 512; ++i) {
    int e = e0 + i;
    int d = (int)dst[e], s = (int)src[e];
    float xi0 = x[d * 3], xi1 = x[d * 3 + 1], xi2 = x[d * 3 + 2];
    float d0 = x[s * 3] - xi0, d1 = x[s * 3 + 1] - xi1, d2 = x[s * 3 + 2] - xi2;
    float v = bias + xi0 * w0 + xi1 * w1 + xi2 * w2 + d0 * w3 + d1 * w4 + d2 * w5;
    v = v > 0.f ? v : 0.f;
    s1 += v; s2 += v * v;
    h1[(size_t)e * 128 + c] = (h16)v;
  }
  atomicAdd(&gsum[c], s1);
  atomicAdd(&gsq[c], s2);
}

// ------------------------------------------ conv2-4 layer1: [E,256]x[256,128]
__global__ __launch_bounds__(256) void k_edge_gemm1(
    const float* __restrict__ xin,            // [N,128]
    const long long* __restrict__ src, const long long* __restrict__ dst,
    const h16* __restrict__ w1t,              // [128][256] W^T f16
    const float* __restrict__ b1,
    h16* __restrict__ h1,                     // [E,128] out
    float* __restrict__ gsum, float* __restrict__ gsq)
{
  __shared__ h16 feat[16 * 264];              // 16 edges x 256(+8 pad) halves
  __shared__ h16 hout[16 * 128];
  __shared__ float lsum[128], lsq[128];
  __shared__ int ld[16], ls[16];
  const int tid = threadIdx.x;
  const int lane = tid & 31, wv = tid >> 5, l16 = lane & 15, hi = lane >> 4;
  if (tid < 128) { lsum[tid] = 0.f; lsq[tid] = 0.f; }
  const int ch = wv * 16 + l16;
  const float bias = b1[ch];
  float s1 = 0.f, s2 = 0.f;
  const int e_base = blockIdx.x * 256;
  const v8f z = {0.f, 0.f, 0.f, 0.f, 0.f, 0.f, 0.f, 0.f};
  const h16* brow = w1t + (size_t)ch * 256;   // loop-invariant: B frags get hoisted
  for (int t = 0; t < 16; ++t) {
    const int e0 = e_base + t * 16;
    __syncthreads();
    if (tid < 16) { ld[tid] = (int)dst[e0 + tid]; ls[tid] = (int)src[e0 + tid]; }
    __syncthreads();
    for (int i = tid; i < 16 * 128; i += 256) {
      int el = i >> 7, cc = i & 127;
      float xi = xin[ld[el] * 128 + cc];
      float xj = xin[ls[el] * 128 + cc];
      feat[el * 264 + cc] = (h16)xi;
      feat[el * 264 + 128 + cc] = (h16)(xj - xi);
    }
    __syncthreads();
    const h16* arow = feat + l16 * 264;
    // issue ALL A-fragment LDS loads before any WMMA
    v16h a[8];
#pragma unroll
    for (int kb = 0; kb < 8; ++kb) a[kb] = frag_a(arow, kb * 32, hi);
    SCHED_FENCE();
    v8f acc = z;
#pragma unroll
    for (int kb = 0; kb < 8; ++kb)
      acc = WMMA(a[kb], frag_b(brow, kb * 32, hi), acc);
#pragma unroll
    for (int r = 0; r < 8; ++r) {
      float v = acc[r] + bias; v = v > 0.f ? v : 0.f;
      s1 += v; s2 += v * v;
      hout[(hi * 8 + r) * 128 + ch] = (h16)v;
    }
    __syncthreads();
    {
      int i = tid * 8, el = i >> 7, cc = i & 127;
      *(uint4*)(&h1[(size_t)(e0 + el) * 128 + cc]) = *(const uint4*)(&hout[i]);
    }
  }
  __syncthreads();
  atomicAdd(&lsum[ch], s1);
  atomicAdd(&lsq[ch], s2);
  __syncthreads();
  if (tid < 128) atomicAdd(&gsum[tid], lsum[tid]);
  else           atomicAdd(&gsq[tid - 128], lsq[tid - 128]);
}

// ----------------- fold BN1 affine into W2/b2 (scale=g*rsqrt(var), shift=be-m*scale)
__global__ __launch_bounds__(128) void k_fold_bn(
    const float* __restrict__ sum, const float* __restrict__ sq, float count,
    const float* __restrict__ g, const float* __restrict__ be,
    const float* __restrict__ W2, const float* __restrict__ b2,
    h16* __restrict__ w2t, float* __restrict__ b2f)
{
  __shared__ float sc[128], sh[128];
  int tid = threadIdx.x;
  {
    float m = sum[tid] / count;
    float v = sq[tid] / count - m * m;
    float s = g[tid] * rsqrtf(v + EPS);
    sc[tid] = s; sh[tid] = be[tid] - m * s;
  }
  __syncthreads();
  float acc = b2[tid];
  for (int k = 0; k < 128; ++k) {
    float wv = W2[k * 128 + tid];
    w2t[tid * 128 + k] = (h16)(sc[k] * wv);
    acc += sh[k] * wv;
  }
  b2f[tid] = acc;
}

// --------------- layer2: [E,128]x[128,128], atomic-max aggregate + BN2 stats
__global__ __launch_bounds__(256) void k_edge_gemm2(
    const h16* __restrict__ h1, const long long* __restrict__ dst,
    const h16* __restrict__ w2t, const float* __restrict__ b2f,
    float* __restrict__ nraw, float* __restrict__ gsum, float* __restrict__ gsq)
{
  __shared__ float lsum[128], lsq[128];
  const int tid = threadIdx.x;
  const int lane = tid & 31, wv = tid >> 5, l16 = lane & 15, hi = lane >> 4;
  if (tid < 128) { lsum[tid] = 0.f; lsq[tid] = 0.f; }
  const int ch = wv * 16 + l16;
  const float bias = b2f[ch];
  float s1 = 0.f, s2 = 0.f;
  const int e_base = blockIdx.x * 256;
  const v8f z = {0.f, 0.f, 0.f, 0.f, 0.f, 0.f, 0.f, 0.f};
  const h16* brow = w2t + (size_t)ch * 128;   // loop-invariant B
  for (int t = 0; t < 16; ++t) {
    const int e0 = e_base + t * 16;
    const h16* arow = h1 + (size_t)(e0 + l16) * 128;
    v16h a[4];
#pragma unroll
    for (int kb = 0; kb < 4; ++kb) a[kb] = frag_a(arow, kb * 32, hi);
    SCHED_FENCE();
    v8f acc = z;
#pragma unroll
    for (int kb = 0; kb < 4; ++kb)
      acc = WMMA(a[kb], frag_b(brow, kb * 32, hi), acc);
#pragma unroll
    for (int r = 0; r < 8; ++r) {
      float v = acc[r] + bias; v = v > 0.f ? v : 0.f;
      s1 += v; s2 += v * v;
      int e = e0 + hi * 8 + r;
      int d = (int)dst[e];
      atomicMax(&nraw[d * 128 + ch], v);
    }
  }
  __syncthreads();
  atomicAdd(&lsum[ch], s1);
  atomicAdd(&lsq[ch], s2);
  __syncthreads();
  if (tid < 128) atomicAdd(&gsum[tid], lsum[tid]);
  else           atomicAdd(&gsq[tid - 128], lsq[tid - 128]);
}

// ------- finalize: BN2 affine applied after max; -inf (no in-edges) -> 0
__global__ __launch_bounds__(256) void k_finalize(
    const float* __restrict__ nraw, const float* __restrict__ sum, const float* __restrict__ sq,
    float count, const float* __restrict__ g, const float* __restrict__ be,
    float* __restrict__ xout, h16* __restrict__ xcat, int col0)
{
  __shared__ float sc[128], sh[128];
  int tid = threadIdx.x;
  if (tid < 128) {
    float m = sum[tid] / count;
    float v = sq[tid] / count - m * m;
    float s = g[tid] * rsqrtf(v + EPS);
    sc[tid] = s; sh[tid] = be[tid] - m * s;
  }
  __syncthreads();
  const int total = Nn * 128;
  for (int i = blockIdx.x * 256 + tid; i < total; i += gridDim.x * 256) {
    int cc = i & 127, nidx = i >> 7;
    float r = nraw[i];
    float val = (r == -__builtin_inff()) ? 0.f : (sc[cc] * r + sh[cc]);
    if (!(val == val)) val = 0.f;
    xout[i] = val;
    xcat[(size_t)nidx * 512 + col0 + cc] = (h16)val;
  }
}

// ----------- lin: [N,512]x[512,512] + ReLU; BN stats + per-graph raw sums
__global__ __launch_bounds__(256) void k_lin_gemm(
    const h16* __restrict__ xcat, const h16* __restrict__ wt, const float* __restrict__ bias,
    const long long* __restrict__ batch,
    float* __restrict__ pooled, float* __restrict__ gsum, float* __restrict__ gsq)
{
  __shared__ float lsum[512], lsq[512];
  const int tid = threadIdx.x;
  const int lane = tid & 31, wv = tid >> 5, l16 = lane & 15, hi = lane >> 4;
  for (int i = tid; i < 512; i += 256) { lsum[i] = 0.f; lsq[i] = 0.f; }
  float s1[4] = {0.f, 0.f, 0.f, 0.f}, s2[4] = {0.f, 0.f, 0.f, 0.f};
  const v8f z = {0.f, 0.f, 0.f, 0.f, 0.f, 0.f, 0.f, 0.f};
  const int m_base = blockIdx.x * 160;                      // 10 tiles x 16 nodes
  for (int t = 0; t < 10; ++t) {
    const int m0 = m_base + t * 16;
    const h16* arow = xcat + (size_t)(m0 + l16) * 512;
    // preload the full K=512 worth of A fragments; global loads issue up front
    v16h a[16];
#pragma unroll
    for (int kb = 0; kb < 16; ++kb) a[kb] = frag_a(arow, kb * 32, hi);
    SCHED_FENCE();
    v8f acc[4] = {z, z, z, z};
    for (int kb = 0; kb < 16; ++kb) {
#pragma unroll
      for (int j = 0; j < 4; ++j) {
        int nrow = (wv + j * 8) * 16 + l16;
        v16h b = frag_b(wt + (size_t)nrow * 512, kb * 32, hi);
        acc[j] = WMMA(a[kb], b, acc[j]);
      }
    }
#pragma unroll
    for (int j = 0; j < 4; ++j) {
      const int ch = (wv + j * 8) * 16 + l16;
      const float bs = bias[ch];
      int gprev = -1; float pacc = 0.f;
#pragma unroll
      for (int r = 0; r < 8; ++r) {
        float v = acc[j][r] + bs; v = v > 0.f ? v : 0.f;
        s1[j] += v; s2[j] += v * v;
        int node = m0 + hi * 8 + r;
        int g = (int)batch[node];                           // batch is sorted -> short runs
        if (g != gprev) {
          if (gprev >= 0) atomicAdd(&pooled[gprev * 512 + ch], pacc);
          pacc = v; gprev = g;
        } else pacc += v;
      }
      atomicAdd(&pooled[gprev * 512 + ch], pacc);
    }
  }
  __syncthreads();
#pragma unroll
  for (int j = 0; j < 4; ++j) {
    int ch = (wv + j * 8) * 16 + l16;
    atomicAdd(&lsum[ch], s1[j]);
    atomicAdd(&lsq[ch], s2[j]);
  }
  __syncthreads();
  for (int i = tid; i < 512; i += 256) {
    atomicAdd(&gsum[i], lsum[i]);
    atomicAdd(&gsq[i], lsq[i]);
  }
}

// ---------------- tail: lin-BN + pool-divide + 2 small MLP layers + head
__global__ __launch_bounds__(256) void k_tail(
    const float* __restrict__ pooled, const float* __restrict__ cnt,
    const float* __restrict__ lsum, const float* __restrict__ lsq,
    const float* __restrict__ gL, const float* __restrict__ beL,
    const float* __restrict__ W1, const float* __restrict__ b1,
    const float* __restrict__ g1, const float* __restrict__ be1,
    const float* __restrict__ W2, const float* __restrict__ b2,
    const float* __restrict__ g2, const float* __restrict__ be2,
    const float* __restrict__ hW, const float* __restrict__ hb,
    float* __restrict__ out, float ncount)
{
  __shared__ float sc[512], sh[512];
  __shared__ float icn[Gb];
  __shared__ float hA[Gb * 128];
  __shared__ float hB[Gb * 64];
  const int tid = threadIdx.x;
  for (int c = tid; c < 512; c += 256) {
    float m = lsum[c] / ncount;
    float v = lsq[c] / ncount - m * m;
    float s = gL[c] * rsqrtf(v + EPS);
    sc[c] = s; sh[c] = beL[c] - m * s;
  }
  if (tid < Gb) { float t = cnt[tid]; icn[tid] = 1.f / (t > 1.f ? t : 1.f); }
  __syncthreads();
  for (int o = tid; o < Gb * 128; o += 256) {
    int g = o >> 7, c = o & 127;
    const float* pg = pooled + g * 512;
    float ic = icn[g];
    float acc = b1[c];
    for (int k = 0; k < 512; ++k)
      acc += (sc[k] * pg[k] * ic + sh[k]) * W1[k * 128 + c];
    hA[o] = acc > 0.f ? acc : 0.f;
  }
  __syncthreads();
  if (tid < 128) {
    float s = 0.f, q = 0.f;
    for (int g = 0; g < Gb; ++g) { float v = hA[g * 128 + tid]; s += v; q += v * v; }
    float m = s / (float)Gb, v = q / (float)Gb - m * m;
    float scv = g1[tid] * rsqrtf(v + EPS);
    sc[tid] = scv; sh[tid] = be1[tid] - m * scv;
  }
  __syncthreads();
  for (int o = tid; o < Gb * 64; o += 256) {
    int g = o >> 6, c = o & 63;
    float acc = b2[c];
    for (int k = 0; k < 128; ++k)
      acc += (sc[k] * hA[g * 128 + k] + sh[k]) * W2[k * 64 + c];
    hB[o] = acc > 0.f ? acc : 0.f;
  }
  __syncthreads();
  if (tid < 64) {
    float s = 0.f, q = 0.f;
    for (int g = 0; g < Gb; ++g) { float v = hB[g * 64 + tid]; s += v; q += v * v; }
    float m = s / (float)Gb, v = q / (float)Gb - m * m;
    float scv = g2[tid] * rsqrtf(v + EPS);
    sc[tid] = scv; sh[tid] = be2[tid] - m * scv;
  }
  __syncthreads();
  for (int o = tid; o < Gb * NCLS; o += 256) {
    int g = o / NCLS, c = o - g * NCLS;
    float acc = hb[c];
    for (int k = 0; k < 64; ++k)
      acc += (sc[k] * hB[g * 64 + k] + sh[k]) * hW[k * NCLS + c];
    out[o] = acc;
  }
}

// ---------------------------------------------------------------- launcher
extern "C" void kernel_launch(void* const* d_in, const int* in_sizes, int n_in,
                              void* d_out, int out_size, void* d_ws, size_t ws_size,
                              hipStream_t stream) {
  (void)in_sizes; (void)n_in; (void)out_size; (void)ws_size;
  const float*     x     = (const float*)d_in[0];
  const long long* ei    = (const long long*)d_in[1];
  const long long* batch = (const long long*)d_in[2];
  const long long* src   = ei;
  const long long* dst   = ei + Ee;
  auto P = [&](int i) { return (const float*)d_in[i]; };

  char* w = (char*)d_ws;
  auto alloc = [&](size_t bytes) { char* p = w; w += (bytes + 255) & ~(size_t)255; return p; };
  h16*   h1     = (h16*)  alloc((size_t)Ee * 128 * 2);     // 164 MB
  h16*   xcat   = (h16*)  alloc((size_t)Nn * 512 * 2);     // 20 MB
  float* xnode  = (float*)alloc((size_t)4 * Nn * 128 * 4); // 41 MB (x1..x4)
  float* nraw   = (float*)alloc((size_t)Nn * 128 * 4);     // 10 MB
  h16*   w1t    = (h16*)  alloc((size_t)3 * 128 * 256 * 2);
  h16*   w2t    = (h16*)  alloc((size_t)128 * 128 * 2);
  h16*   linwt  = (h16*)  alloc((size_t)512 * 512 * 2);
  float* b2f    = (float*)alloc(128 * 4);
  float* cstats = (float*)alloc(512 * 4);
  float* lstats = (float*)alloc(1024 * 4);
  float* pooled = (float*)alloc((size_t)Gb * 512 * 4);
  float* cntb   = (float*)alloc(Gb * 4);
  float *sum1 = cstats, *sq1 = cstats + 128, *sum2 = cstats + 256, *sq2 = cstats + 384;
  float *linsum = lstats, *linsq = lstats + 512;

  // convert W1 (conv2-4) and lin W to transposed f16
  k_convert_wt<<<64, 256, 0, stream>>>(P(11), w1t + 0 * 128 * 256, 256, 128);
  k_convert_wt<<<64, 256, 0, stream>>>(P(19), w1t + 1 * 128 * 256, 256, 128);
  k_convert_wt<<<64, 256, 0, stream>>>(P(27), w1t + 2 * 128 * 256, 256, 128);
  k_convert_wt<<<256, 256, 0, stream>>>(P(35), linwt, 512, 512);

  for (int c = 0; c < 4; ++c) {
    const int base = 3 + c * 8;
    k_fill<<<8, 256, 0, stream>>>(cstats, 512, 0.f);
    k_fill<<<2500, 256, 0, stream>>>(nraw, (long long)Nn * 128, -__builtin_inff());
    if (c == 0) {
      k_edge_conv1_l1<<<1250, 128, 0, stream>>>(x, src, dst, P(3), P(4), h1, sum1, sq1);
    } else {
      const float* xin = xnode + (size_t)(c - 1) * Nn * 128;
      k_edge_gemm1<<<2500, 256, 0, stream>>>(xin, src, dst, w1t + (size_t)(c - 1) * 128 * 256,
                                             P(base + 1), h1, sum1, sq1);
    }
    k_fold_bn<<<1, 128, 0, stream>>>(sum1, sq1, (float)Ee, P(base + 2), P(base + 3),
                                     P(base + 4), P(base + 5), w2t, b2f);
    k_edge_gemm2<<<2500, 256, 0, stream>>>(h1, dst, w2t, b2f, nraw, sum2, sq2);
    k_finalize<<<1250, 256, 0, stream>>>(nraw, sum2, sq2, (float)Ee, P(base + 6), P(base + 7),
                                         xnode + (size_t)c * Nn * 128, xcat, c * 128);
  }

  k_fill<<<4, 256, 0, stream>>>(lstats, 1024, 0.f);
  k_fill<<<64, 256, 0, stream>>>(pooled, (long long)Gb * 512, 0.f);
  k_fill<<<1, 32, 0, stream>>>(cntb, Gb, 0.f);
  k_count<<<40, 256, 0, stream>>>(batch, cntb, Nn);
  k_lin_gemm<<<125, 256, 0, stream>>>(xcat, linwt, P(36), batch, pooled, linsum, linsq);
  k_tail<<<1, 256, 0, stream>>>(pooled, cntb, linsum, linsq, P(37), P(38),
                                P(39), P(40), P(41), P(42),
                                P(43), P(44), P(45), P(46),
                                P(47), P(48), (float*)d_out, (float)Nn);
}
